// spatial_layer_41248865911129
// MI455X (gfx1250) — matrix operand order repaired and checked by reading, not compiled, not run on hardware
//
#include <hip/hip_runtime.h>
#include <hip/hip_bf16.h>

// ---------------------------------------------------------------------------
// CDNA5 (gfx1250): attention block, all GEMMs via v_wmma_f32_16x16x32_bf16.
// 128x128 block tile, 8 waves (wave32), wave tile 32x64 (2x4 WMMA tiles),
// double-buffered LDS. Non-transposed tiles use GLOBAL_LOAD_ASYNC_TO_LDS_B128
// (ASYNCcnt) so copies bypass VGPRs entirely; transposed tiles use VGPR-staged
// scatter. Problem sizes fixed by reference: B=8, N=2048, low=512, high=4096.
// ---------------------------------------------------------------------------

typedef __attribute__((ext_vector_type(16))) __bf16 v16bf;
typedef __attribute__((ext_vector_type(8)))  float  v8f;

#define BIGF 9.0e15f

__device__ __forceinline__ unsigned short f2bf(float f) {
  unsigned int u = __float_as_uint(f);
  unsigned int r = u + 0x7FFFu + ((u >> 16) & 1u);
  return (unsigned short)(r >> 16);
}

// ---- CDNA5 async global->LDS copy (tracked by ASYNCcnt) ----
__device__ __forceinline__ void async_ld_b128(unsigned lds_off, const void* gaddr) {
  asm volatile("global_load_async_to_lds_b128 %0, %1, off"
               :: "v"(lds_off), "v"(gaddr) : "memory");
}
__device__ __forceinline__ void async_ld_b128_o16(unsigned lds_off, const void* gaddr) {
  asm volatile("global_load_async_to_lds_b128 %0, %1, off offset:16"
               :: "v"(lds_off), "v"(gaddr) : "memory");
}
__device__ __forceinline__ void wait_asynccnt0() {
  asm volatile("s_wait_asynccnt 0x0" ::: "memory");
}

// -------------------------- elementwise convert ----------------------------
__global__ void f32_to_bf16_kernel(const float* __restrict__ in,
                                   unsigned short* __restrict__ out, size_t n) {
  size_t i = (size_t)blockIdx.x * blockDim.x + threadIdx.x;
  if (i < n) out[i] = f2bf(in[i]);
}

// ------------------------------- WMMA GEMM ---------------------------------
// C[m,n] = scale * sum_k A'[m,k] * B'[k,n]  (+ bias[n])
//   TA==0: A'[m,k] = A[m*lda + k]      TA==1: A'[m,k] = A[k*lda + m]
//   TB==0: B'[k,n] = Bm[n*ldb + k]     TB==1: B'[k,n] = Bm[k*ldb + n]
// Block: 256 threads (8 waves). Block tile 128x128, K step 32.
// Wave grid 4(M) x 2(N); each wave computes 32x64 = 2x4 WMMA accumulators.
union FragBF { uint4 q[2]; v16bf v; };

#define LDSTRIDE 40   // 32 data + 8 pad bf16 elems (80B rows, 16B aligned)

template <int TA, int TB, int OUT_BF, int BIAS>
__global__ __launch_bounds__(256, 1)
void gemm_wmma_bf16(const unsigned short* __restrict__ A,
                    const unsigned short* __restrict__ Bm,
                    const float* __restrict__ bias,
                    void* __restrict__ Cout,
                    int M, int N, int K, int lda, int ldb, int ldc,
                    float scale,
                    long long sA, long long sB, long long sC) {
  const int tid  = threadIdx.x;
  const int lane = tid & 31;
  const int w    = tid >> 5;          // wave 0..7
  const int wm   = w & 3;             // wave row   (4 x 32 rows)
  const int wn   = w >> 2;            // wave col   (2 x 64 cols)
  const int r    = lane & 15;
  const int hi   = lane >> 4;

  const int n0 = blockIdx.x * 128;
  const int m0 = blockIdx.y * 128;
  const int b  = blockIdx.z;
  A  += (size_t)b * sA;
  Bm += (size_t)b * sB;

  __shared__ __align__(16) unsigned short As[2][128][LDSTRIDE]; // As[buf][m][k]
  __shared__ __align__(16) unsigned short Bs[2][128][LDSTRIDE]; // Bs[buf][n][k] (B^T)

  v8f acc[2][4] = {};

  constexpr int NASY = (TA == 0 ? 1 : 0) + (TB == 0 ? 1 : 0);

  // ---- per-thread copy source addressing ----
  auto aPtr = [&](int kk) -> const unsigned short* {
    if (TA == 0)  // thread t: row = t>>1 (128 rows), half = t&1 (2 x 16 k)
      return A + (size_t)(m0 + (tid >> 1)) * lda + kk + (tid & 1) * 16;
    else          // thread t: k = t>>3 (32 k-rows), seg = t&7 (8 x 16 m)
      return A + (size_t)(kk + (tid >> 3)) * lda + m0 + (tid & 7) * 16;
  };
  auto bPtr = [&](int kk) -> const unsigned short* {
    if (TB == 0)
      return Bm + (size_t)(n0 + (tid >> 1)) * ldb + kk + (tid & 1) * 16;
    else
      return Bm + (size_t)(kk + (tid >> 3)) * ldb + n0 + (tid & 7) * 16;
  };

  // ---- async (no VGPR staging) copies for non-transposed tiles ----
  auto asyncA = [&](int buf, int kk) {
    unsigned l = (unsigned)(size_t)&As[buf][tid >> 1][(tid & 1) * 16];
    const void* g = aPtr(kk);
    async_ld_b128(l, g);
    async_ld_b128_o16(l, g);          // IOFFSET applies to both LDS and mem addr
  };
  auto asyncB = [&](int buf, int kk) {
    unsigned l = (unsigned)(size_t)&Bs[buf][tid >> 1][(tid & 1) * 16];
    const void* g = bPtr(kk);
    async_ld_b128(l, g);
    async_ld_b128_o16(l, g);
  };

  // ---- VGPR-staged transpose copies for transposed tiles ----
  uint4 ra[2], rb[2];
  auto loadA = [&](int kk) {
    const uint4* s = (const uint4*)aPtr(kk);
    ra[0] = s[0]; ra[1] = s[1];
  };
  auto loadB = [&](int kk) {
    const uint4* s = (const uint4*)bPtr(kk);
    rb[0] = s[0]; rb[1] = s[1];
  };
  auto storeA = [&](int buf) {
    union { uint4 q[2]; unsigned short u[16]; } t;
    t.q[0] = ra[0]; t.q[1] = ra[1];
    const int k = tid >> 3, seg = tid & 7;
#pragma unroll
    for (int j = 0; j < 16; ++j) As[buf][seg * 16 + j][k] = t.u[j];
  };
  auto storeB = [&](int buf) {
    union { uint4 q[2]; unsigned short u[16]; } t;
    t.q[0] = rb[0]; t.q[1] = rb[1];
    const int k = tid >> 3, seg = tid & 7;
#pragma unroll
    for (int j = 0; j < 16; ++j) Bs[buf][seg * 16 + j][k] = t.u[j];
  };

  auto compute = [&](int buf) {
    FragBF fa[2], fb[4];
#pragma unroll
    for (int mi = 0; mi < 2; ++mi) {
      const unsigned short* p = &As[buf][wm * 32 + mi * 16 + r][0];
      fa[mi].q[0] = *(const uint4*)(p + hi * 8);
      fa[mi].q[1] = *(const uint4*)(p + 16 + hi * 8);
    }
#pragma unroll
    for (int ni = 0; ni < 4; ++ni) {
      const unsigned short* p = &Bs[buf][wn * 64 + ni * 16 + r][0];
      fb[ni].q[0] = *(const uint4*)(p + hi * 8);
      fb[ni].q[1] = *(const uint4*)(p + 16 + hi * 8);
    }
#pragma unroll
    for (int mi = 0; mi < 2; ++mi)
#pragma unroll
      for (int ni = 0; ni < 4; ++ni)
        acc[mi][ni] = __builtin_amdgcn_wmma_f32_16x16x32_bf16(
            false, fa[mi].v, false, fb[ni].v, (short)0, acc[mi][ni], false, false);
  };

  // ---- software-pipelined main loop (double-buffered LDS) ----
  const int nk = K / 32;

  if (TA == 0) asyncA(0, 0); else { loadA(0); storeA(0); }
  if (TB == 0) asyncB(0, 0); else { loadB(0); storeB(0); }
  if (NASY > 0) wait_asynccnt0();
  __syncthreads();

  for (int kt = 0; kt < nk; ++kt) {
    const int cur = kt & 1;
    const bool haveNext = (kt + 1 < nk);
    if (haveNext) {
      if (TA == 0) asyncA(cur ^ 1, (kt + 1) * 32); else loadA((kt + 1) * 32);
      if (TB == 0) asyncB(cur ^ 1, (kt + 1) * 32); else loadB((kt + 1) * 32);
    }
    if (kt + 2 < nk) {
      __builtin_prefetch(aPtr((kt + 2) * 32), 0, 1);
      __builtin_prefetch(bPtr((kt + 2) * 32), 0, 1);
    }
    compute(cur);                      // 8x wmma hides copy latency
    if (TA != 0 || TB != 0) {
      __syncthreads();                 // everyone done reading buf `cur`
      if (haveNext) {
        if (TA != 0) storeA(cur ^ 1);
        if (TB != 0) storeB(cur ^ 1);
      }
    }
    if (NASY > 0 && haveNext) wait_asynccnt0();  // my async tile kt+1 landed
    __syncthreads();                   // buf cur^1 fully written by all waves
  }

  // ---- epilogue: vgpr i -> row (hi*8 + i), col r within each 16x16 tile ----
#pragma unroll
  for (int mi = 0; mi < 2; ++mi) {
#pragma unroll
    for (int ni = 0; ni < 4; ++ni) {
      const int cn = n0 + wn * 64 + ni * 16 + r;
      const float bv = BIAS ? bias[cn] : 0.0f;
#pragma unroll
      for (int i = 0; i < 8; ++i) {
        const int cm = m0 + wm * 32 + mi * 16 + hi * 8 + i;
        const float val = acc[mi][ni][i] * scale + bv;
        if (OUT_BF)
          ((unsigned short*)Cout)[(size_t)b * sC + (size_t)cm * ldc + cn] = f2bf(val);
        else
          ((float*)Cout)[(size_t)b * sC + (size_t)cm * ldc + cn] = val;
      }
    }
  }
}

// ------------------ fused global+local softmax over e rows -----------------
__device__ __forceinline__ float blockRedMax(float v, float* red) {
  int tid = threadIdx.x;
  red[tid] = v; __syncthreads();
  for (int s = 128; s > 0; s >>= 1) {
    if (tid < s) red[tid] = fmaxf(red[tid], red[tid + s]);
    __syncthreads();
  }
  float r = red[0]; __syncthreads();
  return r;
}
__device__ __forceinline__ float blockRedMin(float v, float* red) {
  int tid = threadIdx.x;
  red[tid] = v; __syncthreads();
  for (int s = 128; s > 0; s >>= 1) {
    if (tid < s) red[tid] = fminf(red[tid], red[tid + s]);
    __syncthreads();
  }
  float r = red[0]; __syncthreads();
  return r;
}
__device__ __forceinline__ float blockRedSum(float v, float* red) {
  int tid = threadIdx.x;
  red[tid] = v; __syncthreads();
  for (int s = 128; s > 0; s >>= 1) {
    if (tid < s) red[tid] += red[tid + s];
    __syncthreads();
  }
  float r = red[0]; __syncthreads();
  return r;
}

// grid: (low, B); block: 256. attn = softmax(e) + softmax(minmax_masked(e)) (bf16).
__global__ __launch_bounds__(256)
void softmax_combine_kernel(const float* __restrict__ e,
                            const int* __restrict__ linkage,
                            unsigned short* __restrict__ attn,
                            int high) {
  const int l = blockIdx.x;
  const int b = blockIdx.y;
  const int low = gridDim.x;
  const float* erow = e + ((size_t)b * low + l) * high;
  const int* mrow = linkage + (size_t)l * high;
  unsigned short* orow = attn + ((size_t)b * low + l) * high;

  __shared__ float row[4096];
  __shared__ float red[256];

  const int tid = threadIdx.x;
  float gmax = -INFINITY, mmin = BIGF, mmax = -BIGF;
  for (int h = tid; h < high; h += 256) {
    float v = erow[h];
    row[h] = v;
    gmax = fmaxf(gmax, v);
    if (mrow[h] > 0) { mmin = fminf(mmin, v); mmax = fmaxf(mmax, v); }
  }
  __syncthreads();
  gmax = blockRedMax(gmax, red);
  mmin = blockRedMin(mmin, red);
  mmax = blockRedMax(mmax, red);

  float den = mmax - mmin;
  if (den == 0.0f) den = 1e-6f;
  const float lmax = (mmax > -BIGF) ? (mmax - mmin) / den : -BIGF;

  float gs = 0.0f, ls = 0.0f;
  for (int h = tid; h < high; h += 256) {
    float v = row[h];
    gs += expf(v - gmax);
    float le = (mrow[h] > 0) ? (v - mmin) / den : -BIGF;
    ls += expf(le - lmax);
  }
  gs = blockRedSum(gs, red);
  ls = blockRedSum(ls, red);

  for (int h = tid; h < high; h += 256) {
    float v = row[h];
    float g = expf(v - gmax) / gs;
    float le = (mrow[h] > 0) ? (v - mmin) / den : -BIGF;
    float lo = expf(le - lmax) / ls;
    orow[h] = f2bf(g + lo);
  }
}

// ------------------------------ launcher -----------------------------------
extern "C" void kernel_launch(void* const* d_in, const int* in_sizes, int n_in,
                              void* d_out, int out_size, void* d_ws, size_t ws_size,
                              hipStream_t stream) {
  (void)in_sizes; (void)n_in; (void)out_size; (void)ws_size;

  constexpr int BATCH = 8, NT = 2048, LOW = 512, HIGH = 4096;
  constexpr size_t BN = (size_t)BATCH * NT;   // 16384 flattened rows

  const float* x_in    = (const float*)d_in[0];
  const int*   linkage = (const int*)  d_in[1];
  const float* wq_w = (const float*)d_in[2];
  const float* wq_b = (const float*)d_in[3];
  const float* wk_w = (const float*)d_in[4];
  const float* wk_b = (const float*)d_in[5];
  const float* wv_w = (const float*)d_in[6];
  const float* wv_b = (const float*)d_in[7];
  const float* o_w  = (const float*)d_in[8];
  const float* o_b  = (const float*)d_in[9];
  float* out = (float*)d_out;

  char* ws = (char*)d_ws;
  size_t off = 0;
  auto alloc = [&](size_t bytes) -> char* {
    char* p = ws + off;
    off = (off + bytes + 255) & ~(size_t)255;
    return p;
  };
  unsigned short* inb  = (unsigned short*)alloc(BN * LOW * 2);
  unsigned short* wqb  = (unsigned short*)alloc((size_t)LOW * LOW * 2);
  unsigned short* wkb  = (unsigned short*)alloc((size_t)HIGH * LOW * 2);
  unsigned short* wvb  = (unsigned short*)alloc((size_t)LOW * LOW * 2);
  unsigned short* owb  = (unsigned short*)alloc((size_t)HIGH * HIGH * 2);
  unsigned short* qb   = (unsigned short*)alloc(BN * LOW * 2);
  unsigned short* kb   = (unsigned short*)alloc(BN * HIGH * 2);
  unsigned short* vb   = (unsigned short*)alloc(BN * LOW * 2);
  float*          ebuf = (float*)         alloc((size_t)BATCH * LOW * HIGH * 4);
  unsigned short* attn = (unsigned short*)alloc((size_t)BATCH * LOW * HIGH * 2);
  unsigned short* xb   = (unsigned short*)alloc(BN * HIGH * 2);

  // ---- 1) f32 -> bf16 conversions ----
  auto cvt = [&](const float* src, unsigned short* dst, size_t n) {
    f32_to_bf16_kernel<<<dim3((unsigned)((n + 255) / 256)), dim3(256), 0, stream>>>(src, dst, n);
  };
  cvt(x_in, inb, BN * LOW);
  cvt(wq_w, wqb, (size_t)LOW * LOW);
  cvt(wk_w, wkb, (size_t)HIGH * LOW);
  cvt(wv_w, wvb, (size_t)LOW * LOW);
  cvt(o_w,  owb, (size_t)HIGH * HIGH);

  // ---- 2) projections: Q,K,V = inputs @ W^T + b  (NT, bf16 out) ----
  gemm_wmma_bf16<0, 0, 1, 1><<<dim3(LOW / 128, BN / 128, 1), 256, 0, stream>>>(
      inb, wqb, wq_b, qb, (int)BN, LOW, LOW, LOW, LOW, LOW, 1.0f, 0, 0, 0);
  gemm_wmma_bf16<0, 0, 1, 1><<<dim3(HIGH / 128, BN / 128, 1), 256, 0, stream>>>(
      inb, wkb, wk_b, kb, (int)BN, HIGH, LOW, LOW, LOW, HIGH, 1.0f, 0, 0, 0);
  gemm_wmma_bf16<0, 0, 1, 1><<<dim3(LOW / 128, BN / 128, 1), 256, 0, stream>>>(
      inb, wvb, wv_b, vb, (int)BN, LOW, LOW, LOW, LOW, LOW, 1.0f, 0, 0, 0);

  // ---- 3) e[b] = (Q[b]^T @ K[b]) / sqrt(high)  (TN, f32 out, batched) ----
  gemm_wmma_bf16<1, 1, 0, 0><<<dim3(HIGH / 128, LOW / 128, BATCH), 256, 0, stream>>>(
      qb, kb, nullptr, ebuf, LOW, HIGH, NT, LOW, HIGH, HIGH, 1.0f / 64.0f,
      (long long)NT * LOW, (long long)NT * HIGH, (long long)LOW * HIGH);

  // ---- 4) attn = softmax(e) + softmax(minmax_masked(e))  (bf16) ----
  softmax_combine_kernel<<<dim3(LOW, BATCH), 256, 0, stream>>>(ebuf, linkage, attn, HIGH);

  // ---- 5) x[b] = V[b] @ attn[b]  (NN, bf16 out, batched) ----
  gemm_wmma_bf16<0, 1, 1, 0><<<dim3(HIGH / 128, NT / 128, BATCH), 256, 0, stream>>>(
      vb, attn, nullptr, xb, NT, HIGH, LOW, LOW, HIGH, HIGH, 1.0f,
      (long long)NT * LOW, (long long)LOW * HIGH, (long long)NT * HIGH);

  // ---- 6) out = x @ o_w^T + o_b  (NT, f32 out into d_out) ----
  gemm_wmma_bf16<0, 0, 0, 1><<<dim3(HIGH / 128, BN / 128, 1), 256, 0, stream>>>(
      xb, owb, o_b, out, (int)BN, HIGH, HIGH, HIGH, HIGH, HIGH, 1.0f, 0, 0, 0);
}
